// GIN_57638461112633
// MI455X (gfx1250) — matrix operand order repaired and checked by reading, not compiled, not run on hardware
//
#include <hip/hip_runtime.h>
#include <hip/hip_bf16.h>
#include <math.h>

#define HID 128

typedef __attribute__((ext_vector_type(16))) __bf16 v16bf;
typedef __attribute__((ext_vector_type(8)))  float  v8f;

// ---------------------------------------------------------------------------
// Edge scatter-add: agg[dst][:] += x[src][:]   (one edge per 32 lanes,
// each lane owns 4 consecutive feats -> float4 gather + 4 f32 atomics)
// ---------------------------------------------------------------------------
__global__ __launch_bounds__(256) void gin_scatter_kernel(
    const float* __restrict__ x, const int* __restrict__ ei,
    float* __restrict__ agg, int nEdges)
{
    long t = (long)blockIdx.x * blockDim.x + threadIdx.x;
    int e = (int)(t >> 5);
    int lane = (int)(t & 31);
    if (e >= nEdges) return;
    int src = ei[e];
    int dst = ei[nEdges + e];
    const float4 v = *(const float4*)(x + (size_t)src * HID + lane * 4);
    float* d = agg + (size_t)dst * HID + lane * 4;
    atomicAdd(d + 0, v.x);
    atomicAdd(d + 1, v.y);
    atomicAdd(d + 2, v.z);
    atomicAdd(d + 3, v.w);
}

// ---------------------------------------------------------------------------
// Fused GIN MLP: out = relu( relu(BN((cur+agg) @ W1 + b1)) @ W2 + b2 )
// 256 threads = 8 waves, each wave: 16 nodes x 128 out via WMMA bf16.
// Block covers 128 nodes (grid padded; out buffer has padded rows so all
// stores are unconditional; load row index clamped so x reads stay in range).
// W1^T / W2^T staged in LDS as bf16 [n][k]; the per-wave 16x128 intermediate
// activation aliases the W1^T region after GEMM1 (total LDS = 64 KB).
// ---------------------------------------------------------------------------
__global__ __launch_bounds__(256) void gin_mlp_kernel(
    const float* __restrict__ cur, const float* __restrict__ agg,
    const float* __restrict__ w1,  const float* __restrict__ b1,
    const float* __restrict__ gg,  const float* __restrict__ be,
    const float* __restrict__ bm,  const float* __restrict__ bv,
    const float* __restrict__ w2,  const float* __restrict__ b2,
    float* __restrict__ out, int nNodes)
{
    __shared__ __bf16 w1T[HID * HID];   // 32 KB (reused as stage after GEMM1)
    __shared__ __bf16 w2T[HID * HID];   // 32 KB

    const int tid = threadIdx.x;

    // ---- cooperative preload of weights, transposed to [n][k], bf16 ----
    for (int j = 0; j < 64; ++j) {
        int idx = tid + j * 256;           // linear over [k][n] row-major
        int k = idx >> 7, n = idx & 127;
        w1T[n * HID + k] = (__bf16)w1[idx];
        w2T[n * HID + k] = (__bf16)w2[idx];
    }
    __syncthreads();

    const int wave = tid >> 5;
    const int lane = tid & 31;
    const int half = lane >> 4;     // 0: lanes 0-15, 1: lanes 16-31
    const int lq   = lane & 15;
    const long rowBase = (long)blockIdx.x * 128 + wave * 16;

    const v8f vzero = {0.f, 0.f, 0.f, 0.f, 0.f, 0.f, 0.f, 0.f};
    v8f acc[8];
#pragma unroll
    for (int t = 0; t < 8; ++t) acc[t] = vzero;

    // A-frag row for this lane; clamp load index (garbage only pollutes the
    // same output row, which lands in the padded region of `out`).
    const long rowA = rowBase + lq;
    const size_t rowOff = (size_t)(rowA < nNodes ? rowA : 0) * HID;
    const float* cp = cur + rowOff;
    const float* ap = agg + rowOff;

    // ---- GEMM1: (cur+agg)[16x128] @ W1[128x128], K in 4 steps of 32 ----
#pragma unroll
    for (int ks = 0; ks < 4; ++ks) {
        const int k0 = 32 * ks + 8 * half;
        // ISA 16-bit A 16x32 layout: elements 0..7 -> K=k0..k0+7,
        //                            elements 8..15 -> K=k0+16..k0+23
        float4 c0 = *(const float4*)(cp + k0);
        float4 c1 = *(const float4*)(cp + k0 + 4);
        float4 c2 = *(const float4*)(cp + k0 + 16);
        float4 c3 = *(const float4*)(cp + k0 + 20);
        float4 g0 = *(const float4*)(ap + k0);
        float4 g1 = *(const float4*)(ap + k0 + 4);
        float4 g2 = *(const float4*)(ap + k0 + 16);
        float4 g3 = *(const float4*)(ap + k0 + 20);
        v16bf a;
        a[0]  = (__bf16)(c0.x + g0.x); a[1]  = (__bf16)(c0.y + g0.y);
        a[2]  = (__bf16)(c0.z + g0.z); a[3]  = (__bf16)(c0.w + g0.w);
        a[4]  = (__bf16)(c1.x + g1.x); a[5]  = (__bf16)(c1.y + g1.y);
        a[6]  = (__bf16)(c1.z + g1.z); a[7]  = (__bf16)(c1.w + g1.w);
        a[8]  = (__bf16)(c2.x + g2.x); a[9]  = (__bf16)(c2.y + g2.y);
        a[10] = (__bf16)(c2.z + g2.z); a[11] = (__bf16)(c2.w + g2.w);
        a[12] = (__bf16)(c3.x + g3.x); a[13] = (__bf16)(c3.y + g3.y);
        a[14] = (__bf16)(c3.z + g3.z); a[15] = (__bf16)(c3.w + g3.w);
#pragma unroll
        for (int t = 0; t < 8; ++t) {
            // B 32x16 layout: lane n=lane%16, element e -> K = 8*half + e (e<8), 16 + 8*half + (e-8) (e>=8)
            v16bf b;
            const __bf16* wp = &w1T[(t * 16 + lq) * HID + 32 * ks + 8 * half];
#pragma unroll
            for (int e = 0; e < 16; ++e) b[e] = wp[e + (e & 8)];
            acc[t] = __builtin_amdgcn_wmma_f32_16x16x32_bf16(
                false, a, false, b, (short)0, acc[t], false, false);
        }
    }

    __syncthreads();   // everyone done reading W1^T; reuse region as stage

    // ---- BN (folded with b1) + relu, stage to LDS in [m][k] bf16 ----
    __bf16* st = &w1T[wave * 16 * HID];   // per-wave 16x128 tile
#pragma unroll
    for (int t = 0; t < 8; ++t) {
        int c = t * 16 + lq;                       // output channel
        float s  = gg[c] * rsqrtf(bv[c] + 1e-5f);
        float tb = s * b1[c] + be[c] - bm[c] * s;
#pragma unroll
        for (int vv = 0; vv < 8; ++vv) {
            // C layout: VGPR vv -> row m = vv + 8*half, col = lane%16
            float val = fmaxf(acc[t][vv] * s + tb, 0.f);
            st[(vv + 8 * half) * HID + c] = (__bf16)val;
        }
    }
    __syncthreads();

    // ---- GEMM2: stage[16x128] @ W2[128x128] ----
#pragma unroll
    for (int t = 0; t < 8; ++t) acc[t] = vzero;
#pragma unroll
    for (int ks = 0; ks < 4; ++ks) {
        v16bf a;
        const __bf16* sp = st + lq * HID + 32 * ks + 8 * half;
#pragma unroll
        for (int e = 0; e < 8; ++e) a[e] = sp[e];          // K=k0..k0+7
#pragma unroll
        for (int e = 0; e < 8; ++e) a[8 + e] = sp[16 + e]; // K=k0+16..k0+23
#pragma unroll
        for (int t = 0; t < 8; ++t) {
            v16bf b;
            const __bf16* wp = &w2T[(t * 16 + lq) * HID + 32 * ks + 8 * half];
#pragma unroll
            for (int e = 0; e < 16; ++e) b[e] = wp[e + (e & 8)];
            acc[t] = __builtin_amdgcn_wmma_f32_16x16x32_bf16(
                false, a, false, b, (short)0, acc[t], false, false);
        }
    }

    // ---- +b2, relu, store (unconditional; out has padded rows) ----
#pragma unroll
    for (int t = 0; t < 8; ++t) {
        int c = t * 16 + lq;
        float bb = b2[c];
#pragma unroll
        for (int vv = 0; vv < 8; ++vv) {
            long row = rowBase + vv + 8 * half;
            out[(size_t)row * HID + c] = fmaxf(acc[t][vv] + bb, 0.f);
        }
    }
}

// ---------------------------------------------------------------------------
// global_add_pool: pooled[batch[n]][:] += h[n][:]
// ---------------------------------------------------------------------------
__global__ __launch_bounds__(256) void gin_pool_kernel(
    const float* __restrict__ h, const int* __restrict__ batch,
    float* __restrict__ pooled, int nNodes)
{
    long t = (long)blockIdx.x * blockDim.x + threadIdx.x;
    int node = (int)(t >> 5);
    int lane = (int)(t & 31);
    if (node >= nNodes) return;
    int g = batch[node];
    const float4 v = *(const float4*)(h + (size_t)node * HID + lane * 4);
    float* d = pooled + (size_t)g * HID + lane * 4;
    atomicAdd(d + 0, v.x);
    atomicAdd(d + 1, v.y);
    atomicAdd(d + 2, v.z);
    atomicAdd(d + 3, v.w);
}

// ph[g][j] = relu(sum_k pooled[g][k] * w[k][j] + b[j])    (256 x 128)
__global__ __launch_bounds__(128) void gin_head1_kernel(
    const float* __restrict__ pooled, const float* __restrict__ w,
    const float* __restrict__ b, float* __restrict__ ph)
{
    int g = blockIdx.x, j = threadIdx.x;
    float acc = b[j];
    for (int k = 0; k < HID; ++k)
        acc += pooled[g * HID + k] * w[k * HID + j];
    ph[g * HID + j] = fmaxf(acc, 0.f);
}

// out[g] = sum_j ph[g][j] * w2[j] + b2
__global__ __launch_bounds__(256) void gin_head2_kernel(
    const float* __restrict__ ph, const float* __restrict__ w2,
    const float* __restrict__ b2, float* __restrict__ out, int nGraphs)
{
    int g = blockIdx.x * blockDim.x + threadIdx.x;
    if (g >= nGraphs) return;
    float acc = b2[0];
    for (int j = 0; j < HID; ++j)
        acc += ph[g * HID + j] * w2[j];
    out[g] = acc;
}

// ---------------------------------------------------------------------------
extern "C" void kernel_launch(void* const* d_in, const int* in_sizes, int n_in,
                              void* d_out, int out_size, void* d_ws, size_t ws_size,
                              hipStream_t stream) {
    const float* x     = (const float*)d_in[0];
    const int*   ei    = (const int*)  d_in[1];
    const int*   batch = (const int*)  d_in[2];
    const int nNodes = in_sizes[0] / HID;
    const int nEdges = in_sizes[1] / 2;
    const int nGraphs = out_size;
    const int nPad = (nNodes + 127) & ~127;   // padded rows for guard-free MLP

    // params flattened in dict order: conv{1,2,3}.{w1,b1,g,be,m,v,w2,b2}, lin1_w, lin1_b, lin2_w, lin2_b
    const float* P[3][8];
    for (int l = 0; l < 3; ++l)
        for (int j = 0; j < 8; ++j)
            P[l][j] = (const float*)d_in[3 + l * 8 + j];
    const float* lin1_w = (const float*)d_in[27];
    const float* lin1_b = (const float*)d_in[28];
    const float* lin2_w = (const float*)d_in[29];
    const float* lin2_b = (const float*)d_in[30];

    float* bufA   = (float*)d_ws;
    float* bufB   = bufA + (size_t)nPad * HID;
    float* pooled = bufB + (size_t)nPad * HID;
    float* ph     = pooled + (size_t)nGraphs * HID;

    const size_t featBytes = (size_t)nPad * HID * sizeof(float);
    const int mlpBlocks = nPad / 128;
    const long scatThreads = (long)nEdges * 32;
    const int scatBlocks = (int)((scatThreads + 255) / 256);
    const long poolThreads = (long)nNodes * 32;
    const int poolBlocks = (int)((poolThreads + 255) / 256);

    // layer 1: agg in bufA, h1 -> bufA (safe: each wave reads its own rows first)
    hipMemsetAsync(bufA, 0, featBytes, stream);
    gin_scatter_kernel<<<scatBlocks, 256, 0, stream>>>(x, ei, bufA, nEdges);
    gin_mlp_kernel<<<mlpBlocks, 256, 0, stream>>>(
        x, bufA, P[0][0], P[0][1], P[0][2], P[0][3], P[0][4], P[0][5],
        P[0][6], P[0][7], bufA, nNodes);

    // layer 2: agg in bufB, h2 -> bufB
    hipMemsetAsync(bufB, 0, featBytes, stream);
    gin_scatter_kernel<<<scatBlocks, 256, 0, stream>>>(bufA, ei, bufB, nEdges);
    gin_mlp_kernel<<<mlpBlocks, 256, 0, stream>>>(
        bufA, bufB, P[1][0], P[1][1], P[1][2], P[1][3], P[1][4], P[1][5],
        P[1][6], P[1][7], bufB, nNodes);

    // layer 3: agg in bufA, h3 -> bufA
    hipMemsetAsync(bufA, 0, featBytes, stream);
    gin_scatter_kernel<<<scatBlocks, 256, 0, stream>>>(bufB, ei, bufA, nEdges);
    gin_mlp_kernel<<<mlpBlocks, 256, 0, stream>>>(
        bufB, bufA, P[2][0], P[2][1], P[2][2], P[2][3], P[2][4], P[2][5],
        P[2][6], P[2][7], bufA, nNodes);

    // pool + head
    hipMemsetAsync(pooled, 0, (size_t)nGraphs * HID * sizeof(float), stream);
    gin_pool_kernel<<<poolBlocks, 256, 0, stream>>>(bufA, batch, pooled, nNodes);
    gin_head1_kernel<<<nGraphs, 128, 0, stream>>>(pooled, lin1_w, lin1_b, ph);
    gin_head2_kernel<<<(nGraphs + 255) / 256, 256, 0, stream>>>(
        ph, lin2_w, lin2_b, (float*)d_out, nGraphs);
}
